// VisionMambaEvaluator_26276609917177
// MI455X (gfx1250) — matrix operand order repaired and checked by reading, not compiled, not run on hardware
//
#include <hip/hip_runtime.h>
#include <hip/hip_bf16.h>

// ---------------------------------------------------------------------------
// Vision Mamba forward -> scalar mean, CDNA5 (gfx1250, wave32, WMMA bf16)
// ---------------------------------------------------------------------------
// L=4096 tokens, D_MODEL=768, D_INNER=1536, D_STATE=16, D_CONV=4, DT_RANK=48.
// GEMMs run as bf16 WMMA (v_wmma_f32_16x16x32_bf16), each wave computing a
// 32x32 register-blocked tile (2x2 accumulators share A/B fragments).
// out_proj GEMM is collapsed to a dot product via linearity of mean().

#define L_TOK   4096
#define DMODEL  768
#define DINNER  1536
#define DSTATE  16
#define DTRANK  48
#define DTPAD   64
#define NBPAD   128   // x_proj output cols padded 80 -> 128

typedef __attribute__((ext_vector_type(16))) __bf16          v16bf;
typedef __attribute__((ext_vector_type(16))) unsigned short  v16u;
typedef __attribute__((ext_vector_type(8)))  float           v8f;

// ---------------- helpers ----------------

__device__ __forceinline__ unsigned short f2bf(float f) {
    unsigned int u = __float_as_uint(f);
    unsigned int r = u + 0x7FFFu + ((u >> 16) & 1u);   // round-nearest-even
    return (unsigned short)(r >> 16);
}

__device__ __forceinline__ float silu(float x) {
    return x / (1.f + __expf(-x));
}

// A fragment: 16x32 (MxK) bf16, row-major source, lda = K stride.
// ISA 7.12.2: lane = 16*half + m ; VGPR v pair -> K = (v<4?0:16) + half*8 + (v&3)*2
__device__ __forceinline__ v16bf load_a_frag(const unsigned short* A, int lda,
                                             int row0, int k0) {
    int lane = threadIdx.x & 31;
    int m = lane & 15, half = lane >> 4;
    const unsigned short* p = A + (size_t)(row0 + m) * lda + k0;
    v16u au;
#pragma unroll
    for (int v = 0; v < 8; ++v) {
        int kb = ((v < 4) ? 0 : 16) + half * 8 + (v & 3) * 2;
        au[2 * v]     = p[kb];
        au[2 * v + 1] = p[kb + 1];
    }
    return __builtin_bit_cast(v16bf, au);
}

// B fragment: 32x16 (KxN) where source W is (N x K) row-major (weight layout),
// i.e. B[k][n] = W[col0+n][k0+k].  element s: K = 8*half + s (s<8), 16 + 8*half + (s-8) (s>=8),
// lane holds column n.  Per-lane read is two runs of 8 contiguous bf16 -> b128 loads.
__device__ __forceinline__ v16bf load_b_frag(const unsigned short* W, int ldw,
                                             int col0, int k0) {
    int lane = threadIdx.x & 31;
    int n = lane & 15, half = lane >> 4;
    const unsigned short* p = W + (size_t)(col0 + n) * ldw + k0 + half * 8;
    v16u bu;
#pragma unroll
    for (int v = 0; v < 16; ++v) bu[v] = p[v + ((v < 8) ? 0 : 8)];
    return __builtin_bit_cast(v16bf, bu);
}

// C 16x16 f32: VGPR r -> M = r + 8*half, N = lane&15
__device__ __forceinline__ void store_c(float* C, int ldc, int row0, int col0, v8f c) {
    int lane = threadIdx.x & 31;
    int n = lane & 15, half = lane >> 4;
#pragma unroll
    for (int r = 0; r < 8; ++r)
        C[(size_t)(row0 + half * 8 + r) * ldc + col0 + n] = c[r];
}

__device__ __forceinline__ v8f wmma_bf16(v16bf a, v16bf b, v8f c) {
    return __builtin_amdgcn_wmma_f32_16x16x32_bf16(false, a, false, b,
                                                   (short)0, c, false, false);
}

// ---------------- kernels ----------------

// x (1,3,1024,1024) -> u bf16 (4096 x 768); c = ch*256 + r*16 + cc, l = ph*64+pw
__global__ void k_patchify(const float* __restrict__ x, unsigned short* __restrict__ u) {
    int idx = blockIdx.x * blockDim.x + threadIdx.x;
    if (idx >= L_TOK * DMODEL) return;
    int l = idx / DMODEL, c = idx % DMODEL;
    int ph = l >> 6, pw = l & 63;
    int ch = c >> 8, rem = c & 255, r = rem >> 4, cc = rem & 15;
    float v = x[(size_t)ch * 1024 * 1024 + (size_t)(ph * 16 + r) * 1024 + pw * 16 + cc];
    u[idx] = f2bf(v);
}

__global__ void k_f32_to_bf16(const float* __restrict__ in, unsigned short* __restrict__ out, int n) {
    int i = blockIdx.x * blockDim.x + threadIdx.x;
    if (i < n) out[i] = f2bf(in[i]);
}

// rows x in_cols f32 -> rows x out_cols bf16, zero padded on the right (K pad)
__global__ void k_pad_cols_bf16(const float* __restrict__ in, unsigned short* __restrict__ out,
                                int rows, int in_cols, int out_cols) {
    int i = blockIdx.x * blockDim.x + threadIdx.x;
    if (i >= rows * out_cols) return;
    int r = i / out_cols, c = i % out_cols;
    out[i] = (c < in_cols) ? f2bf(in[(size_t)r * in_cols + c]) : (unsigned short)0;
}

// in_rows x cols f32 -> out_rows x cols bf16, zero rows appended (N pad)
__global__ void k_pad_rows_bf16(const float* __restrict__ in, unsigned short* __restrict__ out,
                                int in_rows, int out_rows, int cols) {
    int i = blockIdx.x * blockDim.x + threadIdx.x;
    if (i >= out_rows * cols) return;
    int r = i / cols;
    out[i] = (r < in_rows) ? f2bf(in[i]) : (unsigned short)0;
}

// C(MxN) = A(MxK,bf16) * W(NxK,bf16)^T
// Block = 8 waves (2 M x 4 N), block tile 64x128; each wave: 32x32 via 2x2
// accumulators sharing A/B fragments (2 A-frag + 2 B-frag loads per 4 WMMA).
// Requires M % 64 == 0, N % 128 == 0, K % 32 == 0.
__global__ void k_gemm_bf16(const unsigned short* __restrict__ A,
                            const unsigned short* __restrict__ W,
                            float* __restrict__ C, int M, int N, int K) {
    int wave = threadIdx.x >> 5;
    int row0 = blockIdx.y * 64  + (wave >> 2) * 32;
    int col0 = blockIdx.x * 128 + (wave & 3)  * 32;
    v8f acc00 = {}, acc01 = {}, acc10 = {}, acc11 = {};
    for (int k0 = 0; k0 < K; k0 += 32) {
        v16bf a0 = load_a_frag(A, K, row0,      k0);
        v16bf a1 = load_a_frag(A, K, row0 + 16, k0);
        v16bf b0 = load_b_frag(W, K, col0,      k0);
        v16bf b1 = load_b_frag(W, K, col0 + 16, k0);
        acc00 = wmma_bf16(a0, b0, acc00);
        acc01 = wmma_bf16(a0, b1, acc01);
        acc10 = wmma_bf16(a1, b0, acc10);
        acc11 = wmma_bf16(a1, b1, acc11);
    }
    store_c(C, N, row0,      col0,      acc00);
    store_c(C, N, row0,      col0 + 16, acc01);
    store_c(C, N, row0 + 16, col0,      acc10);
    store_c(C, N, row0 + 16, col0 + 16, acc11);
}

// depthwise causal conv (k=4, pad-left 3) + bias + SiLU on xs = xz[:, :1536];
// emits f32 and bf16 copies
__global__ void k_conv_silu(const float* __restrict__ xz,
                            const float* __restrict__ cw,
                            const float* __restrict__ cb,
                            float* __restrict__ xsc,
                            unsigned short* __restrict__ xscb) {
    int idx = blockIdx.x * blockDim.x + threadIdx.x;
    if (idx >= L_TOK * DINNER) return;
    int l = idx / DINNER, c = idx % DINNER;
    float acc = cb[c];
#pragma unroll
    for (int j = 0; j < 4; ++j) {
        int ll = l - 3 + j;
        if (ll >= 0) acc += cw[c * 4 + j] * xz[(size_t)ll * (2 * DINNER) + c];
    }
    float s = silu(acc);
    xsc[idx]  = s;
    xscb[idx] = f2bf(s);
}

// dt_in bf16 (L x 64): cols 0..47 from dbc (L x NBPAD), rest zero
__global__ void k_dtin(const float* __restrict__ dbc, unsigned short* __restrict__ dtin) {
    int i = blockIdx.x * blockDim.x + threadIdx.x;
    if (i >= L_TOK * DTPAD) return;
    int l = i / DTPAD, c = i % DTPAD;
    dtin[i] = (c < DTRANK) ? f2bf(dbc[(size_t)l * NBPAD + c]) : (unsigned short)0;
}

// dt = softplus(dtl + b[c]) in place
__global__ void k_softplus(float* __restrict__ dt, const float* __restrict__ b) {
    int i = blockIdx.x * blockDim.x + threadIdx.x;
    if (i >= L_TOK * DINNER) return;
    float v = dt[i] + b[i % DINNER];
    dt[i] = (v > 20.f) ? v : log1pf(__expf(v));
}

// wsum[k] = sum_m out_proj_w[m,k]   (768 x 1536)
__global__ void k_wsum(const float* __restrict__ w, float* __restrict__ wsum) {
    int k = blockIdx.x * blockDim.x + threadIdx.x;
    if (k >= DINNER) return;
    float s = 0.f;
    for (int m = 0; m < DMODEL; ++m) s += w[(size_t)m * DINNER + k];
    wsum[k] = s;
}

// SSM scan: one thread per channel, 16 states in registers, serial in L.
// Accumulates part[c] = sum_l y[l,c] (out_proj collapsed via mean-linearity).
__global__ void k_scan(const float* __restrict__ dt,
                       const float* __restrict__ xsc,
                       const float* __restrict__ xz,
                       const float* __restrict__ dbc,
                       const float* __restrict__ A_log,
                       const float* __restrict__ Dvec,
                       float* __restrict__ part) {
    int c = blockIdx.x * blockDim.x + threadIdx.x;
    if (c >= DINNER) return;
    float As[DSTATE], h[DSTATE];
#pragma unroll
    for (int s = 0; s < DSTATE; ++s) { As[s] = -__expf(A_log[c * DSTATE + s]); h[s] = 0.f; }
    float Dc = Dvec[c];
    float acc = 0.f;
    for (int l = 0; l < L_TOK; ++l) {
        if (l + 8 < L_TOK) {
            __builtin_prefetch(dt  + (size_t)(l + 8) * DINNER + c, 0, 0);
            __builtin_prefetch(xsc + (size_t)(l + 8) * DINNER + c, 0, 0);
        }
        float dtv = dt [(size_t)l * DINNER + c];
        float xv  = xsc[(size_t)l * DINNER + c];
        float zv  = xz [(size_t)l * (2 * DINNER) + DINNER + c];
        const float* bc = dbc + (size_t)l * NBPAD;
        float yl = 0.f;
#pragma unroll
        for (int s = 0; s < DSTATE; ++s) {
            float dA = __expf(dtv * As[s]);
            h[s] = dA * h[s] + dtv * bc[DTRANK + s] * xv;
            yl += h[s] * bc[DTRANK + DSTATE + s];
        }
        acc += (yl + xv * Dc) * silu(zv);
    }
    part[c] = acc;
}

__global__ void k_final(const float* __restrict__ part, const float* __restrict__ wsum,
                        float* __restrict__ out) {
    __shared__ float sm[256];
    float s = 0.f;
    for (int c = threadIdx.x; c < DINNER; c += 256) s += part[c] * wsum[c];
    sm[threadIdx.x] = s;
    __syncthreads();
    for (int st = 128; st > 0; st >>= 1) {
        if (threadIdx.x < st) sm[threadIdx.x] += sm[threadIdx.x + st];
        __syncthreads();
    }
    if (threadIdx.x == 0) out[0] = sm[0] / (float)(L_TOK * DMODEL);
}

// ---------------- launch ----------------

extern "C" void kernel_launch(void* const* d_in, const int* in_sizes, int n_in,
                              void* d_out, int out_size, void* d_ws, size_t ws_size,
                              hipStream_t stream) {
    const float* x        = (const float*)d_in[0];
    const float* in_w     = (const float*)d_in[1];   // (3072, 768)
    const float* conv_w   = (const float*)d_in[2];   // (1536, 1, 4)
    const float* conv_b   = (const float*)d_in[3];   // (1536)
    const float* xproj_w  = (const float*)d_in[4];   // (80, 1536)
    const float* dtproj_w = (const float*)d_in[5];   // (1536, 48)
    const float* dtproj_b = (const float*)d_in[6];   // (1536)
    const float* A_log    = (const float*)d_in[7];   // (1536, 16)
    const float* Dvec     = (const float*)d_in[8];   // (1536)
    const float* out_w    = (const float*)d_in[9];   // (768, 1536)
    float* out = (float*)d_out;

    char* ws = (char*)d_ws;
    size_t off = 0;
    auto alloc = [&](size_t bytes) { char* p = ws + off; off = (off + bytes + 255) & ~(size_t)255; return p; };

    unsigned short* u_bf   = (unsigned short*)alloc((size_t)L_TOK  * DMODEL * 2);
    unsigned short* win_bf = (unsigned short*)alloc((size_t)2 * DINNER * DMODEL * 2);
    float*          xz     = (float*)         alloc((size_t)L_TOK  * 2 * DINNER * 4);
    float*          xsc    = (float*)         alloc((size_t)L_TOK  * DINNER * 4);
    unsigned short* xsc_bf = (unsigned short*)alloc((size_t)L_TOK  * DINNER * 2);
    unsigned short* xpw_bf = (unsigned short*)alloc((size_t)NBPAD  * DINNER * 2);
    float*          dbc    = (float*)         alloc((size_t)L_TOK  * NBPAD * 4);
    unsigned short* dtin   = (unsigned short*)alloc((size_t)L_TOK  * DTPAD * 2);
    unsigned short* dtw_bf = (unsigned short*)alloc((size_t)DINNER * DTPAD * 2);
    float*          dt     = (float*)         alloc((size_t)L_TOK  * DINNER * 4);
    float*          wsum   = (float*)         alloc((size_t)DINNER * 4);
    float*          part   = (float*)         alloc((size_t)DINNER * 4);
    (void)ws_size; (void)in_sizes; (void)n_in; (void)out_size;

    const int T = 256;

    // 1. patchify -> u bf16
    k_patchify<<<(L_TOK * DMODEL + T - 1) / T, T, 0, stream>>>(x, u_bf);
    // 2. in_proj weight -> bf16
    k_f32_to_bf16<<<(2 * DINNER * DMODEL + T - 1) / T, T, 0, stream>>>(in_w, win_bf, 2 * DINNER * DMODEL);
    // 3. xz = u @ in_w^T   (4096 x 3072, K=768)
    k_gemm_bf16<<<dim3(2 * DINNER / 128, L_TOK / 64), T, 0, stream>>>(
        u_bf, win_bf, xz, L_TOK, 2 * DINNER, DMODEL);
    // 4. depthwise conv + silu
    k_conv_silu<<<(L_TOK * DINNER + T - 1) / T, T, 0, stream>>>(xz, conv_w, conv_b, xsc, xsc_bf);
    // 5. x_proj weight -> bf16, rows padded 80->128; dbc = xsc @ xproj_w^T (4096 x 128, K=1536)
    k_pad_rows_bf16<<<(NBPAD * DINNER + T - 1) / T, T, 0, stream>>>(xproj_w, xpw_bf, 80, NBPAD, DINNER);
    k_gemm_bf16<<<dim3(NBPAD / 128, L_TOK / 64), T, 0, stream>>>(xsc_bf, xpw_bf, dbc, L_TOK, NBPAD, DINNER);
    // 6. dt GEMM (K padded 48->64): dt = dbc[:, :48] @ dtproj_w^T
    k_dtin<<<(L_TOK * DTPAD + T - 1) / T, T, 0, stream>>>(dbc, dtin);
    k_pad_cols_bf16<<<(DINNER * DTPAD + T - 1) / T, T, 0, stream>>>(dtproj_w, dtw_bf, DINNER, DTRANK, DTPAD);
    k_gemm_bf16<<<dim3(DINNER / 128, L_TOK / 64), T, 0, stream>>>(dtin, dtw_bf, dt, L_TOK, DINNER, DTPAD);
    // 7. dt = softplus(dt + b)
    k_softplus<<<(L_TOK * DINNER + T - 1) / T, T, 0, stream>>>(dt, dtproj_b);
    // 8. collapsed out_proj weights
    k_wsum<<<(DINNER + T - 1) / T, T, 0, stream>>>(out_w, wsum);
    // 9. SSM scan (serial in L, parallel over 1536 channels)
    k_scan<<<DINNER / T, T, 0, stream>>>(dt, xsc, xz, dbc, A_log, Dvec, part);
    // 10. final scalar
    k_final<<<1, T, 0, stream>>>(part, wsum, out);
}
